// Synthesis_8839042695768
// MI455X (gfx1250) — matrix operand order repaired and checked by reading, not compiled, not run on hardware
//
#include <hip/hip_runtime.h>
#include <cmath>

typedef __attribute__((ext_vector_type(16))) __bf16 v16bf;
typedef __attribute__((ext_vector_type(4)))  __bf16 v4bf;
typedef __attribute__((ext_vector_type(8)))  float  v8f;

namespace hw {
constexpr int BATCH=64, STEPS=64, ULEN=64, ENCW=384;
constexpr int H1W=768, H2W=1536, FCWID=2304, MDNW=123;
constexpr int KMIX=10;
constexpr int N1=3072, N2=6144, K1=1536, K2=3072, KF=1536, ACTW=3072;
constexpr int NBLK=64, TPB=256;
constexpr int WVB=TPB/32;           // waves per block (8)
constexpr int NWAVES=NBLK*WVB;      // 512
constexpr int GSTRIDE=NBLK*TPB;

// workspace layout (all segments 256B aligned)
constexpr size_t OFF_W1P=0;                         constexpr size_t SZ_W1P=(size_t)K1*N1*2;
constexpr size_t OFF_W2P=OFF_W1P+SZ_W1P;            constexpr size_t SZ_W2P=(size_t)K2*N2*2;
constexpr size_t OFF_WFP=OFF_W2P+SZ_W2P;            constexpr size_t SZ_WFP=(size_t)KF*FCWID*2;
constexpr size_t OFF_ACT=OFF_WFP+SZ_WFP;            constexpr size_t SZ_ACT=(size_t)BATCH*ACTW*4;
constexpr size_t OFF_Z1 =OFF_ACT+SZ_ACT;            constexpr size_t SZ_Z1 =(size_t)BATCH*N1*4;
constexpr size_t OFF_Z2 =OFF_Z1+SZ_Z1;              constexpr size_t SZ_Z2 =(size_t)BATCH*N2*4;
constexpr size_t OFF_C1 =OFF_Z2+SZ_Z2;              constexpr size_t SZ_C1 =(size_t)BATCH*H1W*4;
constexpr size_t OFF_C2 =OFF_C1+SZ_C1;              constexpr size_t SZ_C2 =(size_t)BATCH*H2W*4;
constexpr size_t OFF_KAP=OFF_C2+SZ_C2;              constexpr size_t SZ_KAP=(size_t)BATCH*KMIX*4;
constexpr size_t OFF_INP=OFF_KAP+SZ_KAP;            constexpr size_t SZ_INP=(size_t)STEPS*BATCH*ENCW*4;
constexpr size_t OFF_ENC=OFF_INP+SZ_INP;            constexpr size_t SZ_ENC=(size_t)BATCH*ULEN*ENCW*4;
constexpr size_t OFF_MSK=OFF_ENC+SZ_ENC;            constexpr size_t SZ_MSK=(size_t)BATCH*ULEN*4;
constexpr size_t OFF_FIN=OFF_MSK+SZ_MSK;            constexpr size_t SZ_FIN=(size_t)BATCH*FCWID*4;
constexpr size_t OFF_M  =OFF_FIN+SZ_FIN;            constexpr size_t SZ_M  =(size_t)BATCH*MDNW*4;
constexpr size_t OFF_BAR=OFF_M+SZ_M;
} // namespace hw

using namespace hw;

static __device__ __forceinline__ float sigmoidf_(float x) { return 1.0f / (1.0f + expf(-x)); }

// ---------------- device-wide sense barrier ----------------
static __device__ void grid_barrier(unsigned* bar) {
  __syncthreads();
  if (threadIdx.x == 0) {
    unsigned g = __hip_atomic_load(&bar[1], __ATOMIC_ACQUIRE, __HIP_MEMORY_SCOPE_AGENT);
    unsigned arrived = __hip_atomic_fetch_add(&bar[0], 1u, __ATOMIC_ACQ_REL, __HIP_MEMORY_SCOPE_AGENT);
    if (arrived == (unsigned)(NBLK - 1)) {
      __hip_atomic_store(&bar[0], 0u, __ATOMIC_RELAXED, __HIP_MEMORY_SCOPE_AGENT);
      __hip_atomic_fetch_add(&bar[1], 1u, __ATOMIC_RELEASE, __HIP_MEMORY_SCOPE_AGENT);
    } else {
      while (__hip_atomic_load(&bar[1], __ATOMIC_ACQUIRE, __HIP_MEMORY_SCOPE_AGENT) == g) {
        __builtin_amdgcn_s_sleep(2);
      }
    }
  }
  __syncthreads();
}

// ---------------- A-fragment staging: act fp32 -> WMMA A layout in LDS (bf16) ----------------
// 2048 bf16 elems per chunk, filled as 512 quads: one b128 load + packed cvt + ds_store_b64 each.
static __device__ __forceinline__ void fill_afrag(__bf16* dst, const float* __restrict__ act,
                                                  int kBase, int kt, int tid) {
  #pragma unroll
  for (int e = tid; e < 512; e += TPB) {
    int mt = e >> 7;               // M tile 0..3
    int r  = e & 127;
    int l  = r >> 2;               // lane 0..31
    int i  = (r & 3) * 4;          // quad start within 16-elem fragment
    int row  = mt * 16 + (l & 15);
    int koff = ((l < 16) ? 0 : 16) + i;  // runs of 4 consecutive K
    const float4 f = *(const float4*)(act + row * ACTW + kBase + kt * 32 + koff);
    v4bf p;
    p[0] = (__bf16)f.x; p[1] = (__bf16)f.y; p[2] = (__bf16)f.z; p[3] = (__bf16)f.w;
    *(v4bf*)(dst + mt * 512 + l * 16 + i) = p;
  }
}

// ---------------- WMMA GEMM stage: z[64,Ntot] = act[:,kBase:kBase+Ktot] @ Wp + bias ----------------
// Wp packed bf16 B-fragments: element ((nt*KT+kt)*512 + lane*16 + i) holds W[kt*32+(lane<16?0:16)+i][nt*16+(lane&15)]
static __device__ void gemm_stage(const __bf16* __restrict__ Wp, const float* __restrict__ bias,
                                  const float* __restrict__ act, float* __restrict__ outZ,
                                  int Ktot, int Ntot, int kBase, bool relu, __bf16* ldsA) {
  const int tid  = threadIdx.x;
  const int lane = tid & 31;
  const int wid  = tid >> 5;
  const int gwave = blockIdx.x * WVB + wid;
  const int NT = Ntot >> 4;
  const int KT = Ktot >> 5;
  const bool active = (gwave < NT);
  v8f acc[4];
  #pragma unroll
  for (int mt = 0; mt < 4; ++mt)
    #pragma unroll
    for (int q = 0; q < 8; ++q) acc[mt][q] = 0.0f;

  __bf16* buf0 = ldsA;
  __bf16* buf1 = ldsA + 2048;
  fill_afrag(buf0, act, kBase, 0, tid);
  __syncthreads();

  for (int kt = 0; kt < KT; ++kt) {
    __bf16* cur = (kt & 1) ? buf1 : buf0;
    __bf16* nxt = (kt & 1) ? buf0 : buf1;
    if (kt + 1 < KT) fill_afrag(nxt, act, kBase, kt + 1, tid);  // overlap with WMMA below
    if (active) {  // wave-uniform branch: EXEC stays all ones for WMMA
      const v16bf* bp = (const v16bf*)(Wp + ((size_t)(gwave * KT + kt) * 512 + lane * 16));
      v16bf bfrag = *bp;
      #pragma unroll
      for (int mt = 0; mt < 4; ++mt) {
        v16bf afrag = *(const v16bf*)(cur + mt * 512 + lane * 16);
        acc[mt] = __builtin_amdgcn_wmma_f32_16x16x32_bf16(
            false, afrag, false, bfrag, (short)0, acc[mt], false, false);
      }
      if (kt + 1 < KT) __builtin_prefetch(Wp + (size_t)(gwave * KT + kt + 1) * 512, 0, 1);
    }
    __syncthreads();  // fills of nxt complete; reads of cur done before it is refilled
  }
  if (active) {
    #pragma unroll
    for (int mt = 0; mt < 4; ++mt) {
      #pragma unroll
      for (int v = 0; v < 8; ++v) {
        int row = mt * 16 + ((lane < 16) ? v : v + 8);  // documented C/D layout
        int col = gwave * 16 + (lane & 15);
        float r = acc[mt][v] + bias[col];
        if (relu) r = fmaxf(r, 0.0f);
        outZ[row * Ntot + col] = r;
      }
    }
  }
}

// ---------------- weight packing (fp32 -> bf16 WMMA B-fragment layout) ----------------
static __device__ void pack_one(const float* __restrict__ Wa, const float* __restrict__ Wb,
                                int splitK, int Ktot, int Ntot, __bf16* __restrict__ P, int gtid) {
  const int KT = Ktot >> 5;
  const int total = Ktot * Ntot;
  for (int e = gtid; e < total; e += GSTRIDE) {
    int tile = e >> 9;
    int r = e & 511;
    int l = r >> 4;
    int i = r & 15;
    int nt = tile / KT;
    int kt = tile - nt * KT;
    int k = kt * 32 + ((l < 16) ? 0 : 16) + i;
    int n = nt * 16 + (l & 15);
    float v = (k < splitK) ? Wa[k * Ntot + n] : Wb[(k - splitK) * Ntot + n];
    P[e] = (__bf16)v;
  }
}

// ---------------- the persistent kernel ----------------
__global__ void __launch_bounds__(TPB, 1)
synth_persistent(const float* y, const float* ctx0, const float* kap0, const float* embed,
                 const float* W_in, const float* b_in,
                 const float* Wx1, const float* Wh1, const float* b1,
                 const float* W_att, const float* b_att,
                 const float* Wx2, const float* Wh2, const float* b2,
                 const float* W_fc, const float* b_fc,
                 const float* W_mdn, const float* b_mdn,
                 const int* x, float* out, char* ws) {
  __shared__ __align__(32) __bf16 ldsA[2 * 2048];  // double-buffered A fragments
  __shared__ float sm[128];

  __bf16* W1p = (__bf16*)(ws + OFF_W1P);
  __bf16* W2p = (__bf16*)(ws + OFF_W2P);
  __bf16* WFp = (__bf16*)(ws + OFF_WFP);
  float* act  = (float*)(ws + OFF_ACT);
  float* z1   = (float*)(ws + OFF_Z1);
  float* z2   = (float*)(ws + OFF_Z2);
  float* c1   = (float*)(ws + OFF_C1);
  float* c2   = (float*)(ws + OFF_C2);
  float* kap  = (float*)(ws + OFF_KAP);
  float* inpA = (float*)(ws + OFF_INP);
  float* enc  = (float*)(ws + OFF_ENC);
  float* msk  = (float*)(ws + OFF_MSK);
  float* fin  = (float*)(ws + OFF_FIN);
  float* mbuf = (float*)(ws + OFF_M);
  unsigned* bar = (unsigned*)(ws + OFF_BAR);

  const int gtid = blockIdx.x * TPB + threadIdx.x;
  const int lane = threadIdx.x & 31;
  const int wid  = threadIdx.x >> 5;
  const int gwave = blockIdx.x * WVB + wid;
  const int b_of_block = blockIdx.x;  // exactly 64 blocks == BATCH

  // ---- phase P: pack weights, precompute inp_t / enc / mask, init states ----
  pack_one(Wx1, Wh1, H1W, K1, N1, W1p, gtid);
  pack_one(Wx2, Wh2, H2W, K2, N2, W2p, gtid);
  pack_one(W_fc, W_fc, KF, KF, FCWID, WFp, gtid);  // splitK==Ktot -> only W_fc used

  for (int e = gtid; e < STEPS * BATCH * ENCW; e += GSTRIDE) {
    int d = e % ENCW;
    int rb = e / ENCW;
    int b = rb % BATCH;
    int t = rb / BATCH;
    const float* yy = y + (b * STEPS + t) * 3;
    float s = b_in[d] + yy[0] * W_in[d] + yy[1] * W_in[ENCW + d] + yy[2] * W_in[2 * ENCW + d];
    inpA[e] = tanhf(s);
  }
  for (int e = gtid; e < BATCH * ULEN * ENCW; e += GSTRIDE) {
    int d = e % ENCW;
    int bu = e / ENCW;
    enc[e] = embed[x[bu] * ENCW + d];
  }
  for (int e = gtid; e < BATCH * ULEN; e += GSTRIDE) msk[e] = (x[e] != 0) ? 1.0f : 0.0f;
  for (int e = gtid; e < BATCH * ACTW; e += GSTRIDE) {
    int d = e % ACTW;
    int b = e / ACTW;
    float v = 0.0f;
    if (d < ENCW) {  // inp at t=0 (recomputed to avoid intra-phase dependency)
      const float* yy = y + (b * STEPS + 0) * 3;
      v = tanhf(b_in[d] + yy[0] * W_in[d] + yy[1] * W_in[ENCW + d] + yy[2] * W_in[2 * ENCW + d]);
    } else if (d < 2 * ENCW) {
      v = ctx0[b * ENCW + (d - ENCW)];
    }
    act[e] = v;  // h1 / h2 slices start at zero
  }
  for (int e = gtid; e < BATCH * H1W; e += GSTRIDE) c1[e] = 0.0f;
  for (int e = gtid; e < BATCH * H2W; e += GSTRIDE) c2[e] = 0.0f;
  for (int e = gtid; e < BATCH * KMIX; e += GSTRIDE) kap[e] = kap0[e];
  grid_barrier(bar);

  // ---- time loop ----
  for (int t = 0; t < STEPS; ++t) {
    // [A] LSTM1 pre-activations: [inp|ctx|h1] (K=1536) x W1 -> z1[64,3072]
    gemm_stage(W1p, b1, act, z1, K1, N1, 0, false, ldsA);
    grid_barrier(bar);

    // [B] LSTM1 gates -> h1 (into act[:,768:1536]), c1
    for (int e = gtid; e < BATCH * H1W; e += GSTRIDE) {
      int j = e % H1W;
      int b = e / H1W;
      const float* z = z1 + b * N1;
      float iv = sigmoidf_(z[j]);
      float fv = sigmoidf_(z[H1W + j]);
      float gv = tanhf(z[2 * H1W + j]);
      float ov = sigmoidf_(z[3 * H1W + j]);
      float c = fv * c1[e] + iv * gv;
      c1[e] = c;
      act[b * ACTW + 768 + j] = ov * tanhf(c);
    }
    grid_barrier(bar);

    // [C] attention: one block per batch row
    {
      int b = b_of_block;
      const float* h1 = act + b * ACTW + 768;
      float* p = sm;         // 30
      float* phi = sm + 32;  // 64
      // wave-parallel dots with shuffle reduction
      for (int j = wid; j < 3 * KMIX; j += WVB) {
        float s = 0.0f;
        for (int k = lane; k < H1W; k += 32) s += h1[k] * W_att[k * 3 * KMIX + j];
        #pragma unroll
        for (int off = 16; off > 0; off >>= 1) s += __shfl_xor(s, off, 32);
        if (lane == 0) p[j] = s + b_att[j];
      }
      __syncthreads();
      if (threadIdx.x < KMIX) kap[b * KMIX + threadIdx.x] += expf(p[2 * KMIX + threadIdx.x]);
      __syncthreads();
      for (int u = threadIdx.x; u < ULEN; u += TPB) {
        float s = 0.0f;
        for (int k = 0; k < KMIX; ++k) {
          float alpha = expf(p[k]);
          float beta = expf(p[KMIX + k]);
          float d = kap[b * KMIX + k] - (float)u;
          s += alpha * expf(-beta * d * d);
        }
        phi[u] = s * msk[b * ULEN + u];
      }
      __syncthreads();
      for (int d = threadIdx.x; d < ENCW; d += TPB) {
        float s = 0.0f;
        for (int u = 0; u < ULEN; ++u) s += phi[u] * enc[(b * ULEN + u) * ENCW + d];
        act[b * ACTW + ENCW + d] = s;  // ctx
      }
    }
    grid_barrier(bar);

    // [D] LSTM2 pre-activations: [inp|ctx|h1|h2] (K=3072) x W2 -> z2[64,6144]
    gemm_stage(W2p, b2, act, z2, K2, N2, 0, false, ldsA);
    grid_barrier(bar);

    // [E] LSTM2 gates -> h2 (into act[:,1536:3072]), c2
    for (int e = gtid; e < BATCH * H2W; e += GSTRIDE) {
      int j = e % H2W;
      int b = e / H2W;
      const float* z = z2 + b * N2;
      float iv = sigmoidf_(z[j]);
      float fv = sigmoidf_(z[H2W + j]);
      float gv = tanhf(z[2 * H2W + j]);
      float ov = sigmoidf_(z[3 * H2W + j]);
      float c = fv * c2[e] + iv * gv;
      c2[e] = c;
      act[b * ACTW + 1536 + j] = ov * tanhf(c);
    }
    grid_barrier(bar);

    // [F] FC with fused ReLU: h2 (act cols 1536..3071) x W_fc -> fin[64,2304]
    gemm_stage(WFp, b_fc, act, fin, KF, FCWID, 1536, true, ldsA);
    grid_barrier(bar);

    // [G] MDN projection: wave-parallel dot per (b,j), lanes stride K, shuffle-reduce
    for (int task = gwave; task < BATCH * MDNW; task += NWAVES) {
      int j = task % MDNW;
      int b = task / MDNW;
      const float* f = fin + b * FCWID;
      float s = 0.0f;
      for (int k = lane; k < FCWID; k += 32) s += f[k] * W_mdn[k * MDNW + j];
      #pragma unroll
      for (int off = 16; off > 0; off >>= 1) s += __shfl_xor(s, off, 32);
      if (lane == 0) mbuf[task] = s + b_mdn[j];
    }
    grid_barrier(bar);

    // [H] MDN transforms + output write; also stage inp_{t+1}
    {
      int b = b_of_block;
      const float* mm = mbuf + b * MDNW;
      float* o = out + (b * STEPS + t) * MDNW;
      if (threadIdx.x == 0) {
        float mx = fmaxf(fmaxf(mm[0], mm[1]), mm[2]);
        float se = expf(mm[0] - mx) + expf(mm[1] - mx) + expf(mm[2] - mx);
        sm[0] = mx + logf(se);  // logsumexp for ce
        float mx2 = mm[3];
        for (int j = 1; j < 20; ++j) mx2 = fmaxf(mx2, mm[3 + j]);
        float s2 = 0.0f;
        for (int j = 0; j < 20; ++j) s2 += expf(mm[3 + j] - mx2);
        sm[1] = mx2;
        sm[2] = s2;
      }
      __syncthreads();
      for (int j = threadIdx.x; j < MDNW; j += TPB) {
        float v;
        if (j < 3) v = mm[j] - sm[0];
        else if (j < 23) v = expf(mm[j] - sm[1]) / sm[2];
        else if (j < 43) v = tanhf(mm[j]);
        else if (j < 83) v = expf(mm[j]);
        else v = mm[j];
        o[j] = v;
      }
      if (t + 1 < STEPS) {
        for (int d = threadIdx.x; d < ENCW; d += TPB)
          act[b * ACTW + d] = inpA[((t + 1) * BATCH + b) * ENCW + d];
      }
    }
    grid_barrier(bar);
  }
}

__global__ void init_barrier(unsigned* bar) {
  if (threadIdx.x < 8) bar[threadIdx.x] = 0u;
}

extern "C" void kernel_launch(void* const* d_in, const int* in_sizes, int n_in,
                              void* d_out, int out_size, void* d_ws, size_t ws_size,
                              hipStream_t stream) {
  const float* y     = (const float*)d_in[0];
  const float* ctx0  = (const float*)d_in[1];
  const float* kap0  = (const float*)d_in[2];
  const float* embed = (const float*)d_in[3];
  const float* W_in  = (const float*)d_in[4];
  const float* b_in  = (const float*)d_in[5];
  const float* Wx1   = (const float*)d_in[6];
  const float* Wh1   = (const float*)d_in[7];
  const float* b1    = (const float*)d_in[8];
  const float* W_att = (const float*)d_in[9];
  const float* b_att = (const float*)d_in[10];
  const float* Wx2   = (const float*)d_in[11];
  const float* Wh2   = (const float*)d_in[12];
  const float* b2    = (const float*)d_in[13];
  const float* W_fc  = (const float*)d_in[14];
  const float* b_fc  = (const float*)d_in[15];
  const float* W_mdn = (const float*)d_in[16];
  const float* b_mdn = (const float*)d_in[17];
  const int*   x     = (const int*)d_in[18];

  char* ws = (char*)d_ws;
  unsigned* bar = (unsigned*)(ws + OFF_BAR);
  init_barrier<<<1, 32, 0, stream>>>(bar);
  synth_persistent<<<NBLK, TPB, 0, stream>>>(
      y, ctx0, kap0, embed, W_in, b_in, Wx1, Wh1, b1, W_att, b_att,
      Wx2, Wh2, b2, W_fc, b_fc, W_mdn, b_mdn, x, (float*)d_out, ws);
}
